// EMAVectorQuantizer_2_32074815767050
// MI455X (gfx1250) — matrix operand order repaired and checked by reading, not compiled, not run on hardware
//
#include <hip/hip_runtime.h>
#include <hip/hip_bf16.h>
#include <math.h>

typedef __attribute__((ext_vector_type(16))) __bf16 v16bf;
typedef __attribute__((ext_vector_type(8)))  float  v8f;

#define N_E    8192
#define E_DIM  256
#define NROWS  32768      // 8*16*16*16
#define SPAT   4096       // 16*16*16
#define BATCH  8
#define DECAYF 0.95f
#define COMMITF 0.25f
#define EPSF   1e-5f

#define AP 264            // LDS A-tile pitch in bf16 elems: 528 B/row, 16-B aligned, 4-dword bank skew

typedef union { v16bf v; uint4 q[2]; } FRAG;

// ---------------- zero scratch ----------------
__global__ void k_zero(float* p, size_t n) {
    size_t i = (size_t)blockIdx.x * blockDim.x + threadIdx.x;
    size_t st = (size_t)gridDim.x * blockDim.x;
    for (; i < n; i += st) p[i] = 0.0f;
}

// ---------------- codebook -> bf16 + row norms ----------------
__global__ void k_prep(const float* __restrict__ cb, __bf16* __restrict__ cbb,
                       float* __restrict__ cbn) {
    const int k = blockIdx.x, t = threadIdx.x;
    float v = cb[(size_t)k * E_DIM + t];
    cbb[(size_t)k * E_DIM + t] = (__bf16)v;
    float s = v * v;
    for (int o = 16; o > 0; o >>= 1) s += __shfl_down(s, o, 32);
    __shared__ float ps[8];
    if ((t & 31) == 0) ps[t >> 5] = s;
    __syncthreads();
    if (t == 0) { float tot = 0.f; for (int i = 0; i < 8; i++) tot += ps[i]; cbn[k] = tot; }
}

// ---------------- helpers for the WMMA GEMM ----------------
__device__ __forceinline__ void load_b(FRAG (&b)[8], const __bf16* __restrict__ brow, int lh) {
#pragma unroll
    for (int kk = 0; kk < 8; kk++) {
        const __bf16* bp = brow + kk * 32 + lh * 8;   // lane half selects K {0..7,16..23} / {8..15,24..31}
        b[kk].q[0] = *(const uint4*)bp;
        b[kk].q[1] = *(const uint4*)(bp + 16);
    }
}

__device__ __forceinline__ void do_chunk(const FRAG (&a)[8], const FRAG (&b)[8],
                                         const float* __restrict__ cbn, int e,
                                         float (&bd)[8], int (&bi)[8]) {
    v8f acc = {};
#pragma unroll
    for (int kk = 0; kk < 8; kk++)
        acc = __builtin_amdgcn_wmma_f32_16x16x32_bf16(
            false, a[kk].v, false, b[kk].v, (short)0, acc, false, false);
    const float nrm = cbn[e];
#pragma unroll
    for (int r = 0; r < 8; r++) {                     // C layout: M = r + 8*lh, N = lane%16
        float d = nrm - 2.0f * acc[r];                // ||c||^2 - 2 x.c (x.x row-const dropped)
        if (d < bd[r] || (d == bd[r] && e < bi[r])) { bd[r] = d; bi[r] = e; }
    }
}

// ---------------- fused distance GEMM (WMMA bf16) + argmin, double-buffered ----------------
__global__ __launch_bounds__(256) void k_argmin(const float* __restrict__ x,
                                                const __bf16* __restrict__ cbb,
                                                const float* __restrict__ cbn,
                                                int* __restrict__ idx_i,
                                                float* __restrict__ idx_f) {
    __shared__ __align__(16) __bf16 la[32 * AP];
    __shared__ float rd[4][32];
    __shared__ int   ri[4][32];

    const int tid = threadIdx.x;
    const int n0  = blockIdx.x * 32;          // 32 x-rows per block
    const int bb  = n0 >> 12;                 // batch index (4096 % 32 == 0)
    const int sp0 = n0 & 4095;

    // transpose-gather x tile into LDS as bf16: la[n_local][c]
    {
        const int nl = tid & 31;
        const float* xb = x + (size_t)bb * ((size_t)E_DIM * SPAT) + sp0 + nl;
        for (int c = tid >> 5; c < E_DIM; c += 8)
            la[nl * AP + c] = (__bf16)xb[(size_t)c * SPAT];
    }
    __syncthreads();

    const int lane = tid & 31, wave = tid >> 5;
    const int rg  = wave >> 2;                // row-group: rows rg*16 .. rg*16+15
    const int col = wave & 3;                 // entry-column within 64-entry chunk
    const int lh  = lane >> 4;                // lane half (A/B/C layout split)
    const int l16 = lane & 15;
    const int m   = rg * 16 + l16;            // A-matrix row for this lane

    // A fragments resident in registers (64 VGPRs), per ISA 16-bit A layout
    FRAG a[8];
#pragma unroll
    for (int kk = 0; kk < 8; kk++) {
        const __bf16* ap = la + m * AP + kk * 32 + lh * 8;
        a[kk].q[0] = *(const uint4*)ap;       // K {0..7} / {8..15}
        a[kk].q[1] = *(const uint4*)(ap + 16);// K {16..23} / {24..31}
    }

    float bd[8];
    int   bi[8];
#pragma unroll
    for (int r = 0; r < 8; r++) { bd[r] = 3.4e38f; bi[r] = 0; }

    // double-buffered sweep: two 64-entry chunks per iteration
    const size_t estride = (size_t)64 * E_DIM;          // bf16 elems between chunks
    const __bf16* brow = cbb + (size_t)(col * 16 + l16) * E_DIM;
    int e = col * 16 + l16;

    FRAG b0[8], b1[8];
    load_b(b0, brow, lh);                                // prologue: chunk 0
    for (int e0 = col * 16; e0 < N_E; e0 += 128) {
        load_b(b1, brow + estride, lh);                  // loads for chunk i+1 in flight
        do_chunk(a, b0, cbn, e, bd, bi);                 // compute chunk i from regs
        if (e0 + 128 < N_E) {
            load_b(b0, brow + 2 * estride, lh);          // loads for chunk i+2 in flight
            __builtin_prefetch((const void*)(brow + 4 * estride), 0, 3);
        }
        do_chunk(a, b1, cbn, e + 64, bd, bi);            // compute chunk i+1 from regs
        brow += 2 * estride;
        e += 128;
    }

    // reduce across the 16 lanes of each half (same M, different entries)
#pragma unroll
    for (int off = 1; off < 16; off <<= 1) {
#pragma unroll
        for (int r = 0; r < 8; r++) {
            float od = __shfl_xor(bd[r], off, 32);
            int   oi = __shfl_xor(bi[r], off, 32);
            if (od < bd[r] || (od == bd[r] && oi < bi[r])) { bd[r] = od; bi[r] = oi; }
        }
    }
    if (l16 == 0) {
#pragma unroll
        for (int r = 0; r < 8; r++) {
            int row = rg * 16 + lh * 8 + r;
            rd[col][row] = bd[r];
            ri[col][row] = bi[r];
        }
    }
    __syncthreads();

    if (tid < 32) {                            // reduce across 4 entry-columns
        float d = rd[0][tid]; int i = ri[0][tid];
        for (int c2 = 1; c2 < 4; c2++) {
            float od = rd[c2][tid]; int oi = ri[c2][tid];
            if (od < d || (od == d && oi < i)) { d = od; i = oi; }
        }
        idx_i[n0 + tid] = i;
        idx_f[n0 + tid] = (float)i;
    }
}

// ---------------- gather + quantized out + segment sums + e_loss ----------------
__global__ void k_scatter(const float* __restrict__ x, const float* __restrict__ cb,
                          const int* __restrict__ idx_i, float* __restrict__ out_q,
                          float* __restrict__ counts, float* __restrict__ dw,
                          float* __restrict__ eacc) {
    const int tid = threadIdx.x;
    const int n0  = blockIdx.x * 32;
    const int bb  = n0 >> 12;
    const int sp0 = n0 & 4095;
    const int nl  = tid & 31;
    const int n   = n0 + nl;
    const int id  = idx_i[n];

    const float* xb = x     + (size_t)bb * ((size_t)E_DIM * SPAT) + sp0 + nl;
    float*       qb = out_q + (size_t)bb * ((size_t)E_DIM * SPAT) + sp0 + nl;

    float s = 0.f;
    for (int c = tid >> 5; c < E_DIM; c += 8) {
        float xv = xb[(size_t)c * SPAT];
        float qv = cb[(size_t)id * E_DIM + c];
        qb[(size_t)c * SPAT] = qv;            // straight-through value == quant
        float df = qv - xv;
        s += df * df;
        atomicAdd(&dw[(size_t)id * E_DIM + c], xv);
    }
    for (int o = 16; o > 0; o >>= 1) s += __shfl_down(s, o, 32);
    __shared__ float ps[8];
    if ((tid & 31) == 0) ps[tid >> 5] = s;
    __syncthreads();
    if (tid == 0) { float t = 0.f; for (int i = 0; i < 8; i++) t += ps[i]; atomicAdd(eacc, t); }
    if (tid < 32) atomicAdd(&counts[idx_i[n0 + tid]], 1.0f);
}

// ---------------- EMA update + perplexity/unique accumulation ----------------
__global__ void k_finalize(const float* __restrict__ ema_count, const float* __restrict__ ema_weight,
                           const float* __restrict__ counts, const float* __restrict__ dw,
                           float* __restrict__ o_newcb, float* __restrict__ o_newcount,
                           float* __restrict__ o_newweight,
                           float* __restrict__ pacc, float* __restrict__ uacc) {
    const int k = blockIdx.x, c = threadIdx.x;
    const float cnt = counts[k];
    float ncnt = ema_count[k] * DECAYF + (1.0f - DECAYF) * cnt;
    ncnt = (ncnt + EPSF) / ((float)BATCH + (float)N_E * EPSF) * (float)BATCH;
    const size_t kc = (size_t)k * E_DIM + c;
    float nw = ema_weight[kc] * DECAYF + (1.0f - DECAYF) * dw[kc];
    o_newweight[kc] = nw;
    o_newcb[kc]     = nw / ncnt;
    if (c == 0) {
        o_newcount[k] = ncnt;
        float p = cnt / (float)NROWS;
        atomicAdd(pacc, p * logf(p + 1e-10f));
        if (cnt != 0.0f) atomicAdd(uacc, 1.0f);
    }
}

__global__ void k_scalars(const float* eacc, const float* pacc, const float* uacc,
                          float* o_eloss, float* o_perp, float* o_uni) {
    *o_eloss = COMMITF * (*eacc) / (float)((size_t)NROWS * E_DIM);
    *o_perp  = expf(-(*pacc));
    *o_uni   = *uacc;
}

// ---------------- launch ----------------
extern "C" void kernel_launch(void* const* d_in, const int* in_sizes, int n_in,
                              void* d_out, int out_size, void* d_ws, size_t ws_size,
                              hipStream_t stream) {
    const float* x    = (const float*)d_in[0];
    const float* cb   = (const float*)d_in[1];
    const float* emc  = (const float*)d_in[2];
    const float* emw  = (const float*)d_in[3];

    // workspace layout (bytes)
    char* ws = (char*)d_ws;
    __bf16* cbb   = (__bf16*)(ws + 0);                        // 4 MB
    float*  cbn   = (float*)(ws + 4194304);                   // 32 KB
    int*    idx_i = (int*)  (ws + 4227072);                   // 128 KB
    float*  cnts  = (float*)(ws + 4358144);                   // 32 KB
    float*  dw    = (float*)(ws + 4390912);                   // 8 MB
    float*  eacc  = (float*)(ws + 12779520);                  // scalars
    float*  pacc  = eacc + 1;
    float*  uacc  = eacc + 2;

    // output layout (floats, tuple return order)
    float* out       = (float*)d_out;
    float* o_quant   = out;                                   // 8388608
    float* o_eloss   = out + 8388608;
    float* o_perp    = out + 8388609;
    float* o_uni     = out + 8388610;
    float* o_newcb   = out + 8388611;                         // 2097152
    float* o_newcnt  = out + 10485763;                        // 8192
    float* o_neww    = out + 10493955;                        // 2097152
    float* o_idx     = out + 12591107;                        // 32768

    // 1) zero accumulators (counts + dw + 3 scalars are contiguous)
    k_zero<<<2048, 256, 0, stream>>>(cnts, (size_t)N_E + (size_t)N_E * E_DIM + 4);
    // 2) codebook bf16 + norms
    k_prep<<<N_E, E_DIM, 0, stream>>>(cb, cbb, cbn);
    // 3) fused WMMA distance + argmin
    k_argmin<<<NROWS / 32, 256, 0, stream>>>(x, cbb, cbn, idx_i, o_idx);
    // 4) gather/quantize + segment sums + e_loss
    k_scatter<<<NROWS / 32, 256, 0, stream>>>(x, cb, idx_i, o_quant, cnts, dw, eacc);
    // 5) EMA + perplexity/unique
    k_finalize<<<N_E, E_DIM, 0, stream>>>(emc, emw, cnts, dw, o_newcb, o_newcnt, o_neww, pacc, uacc);
    // 6) scalars
    k_scalars<<<1, 1, 0, stream>>>(eacc, pacc, uacc, o_eloss, o_perp, o_uni);
}
